// TextEncoder_4681514353165
// MI455X (gfx1250) — compile-verified, matrix operand order checked by reading, not run on hardware
//
#include <hip/hip_runtime.h>
#include <hip/hip_bf16.h>

#define B_   16
#define N_   32
#define D_   512
#define H_   512
#define G4H  2048
#define NT_  16
#define SEM_ 256
#define OUTD 4096
#define SPANS 496              // N*(N-1)/2
#define ROWS  (B_ * SPANS)     // 7936

typedef __attribute__((ext_vector_type(16))) __bf16 v16bf;
typedef __attribute__((ext_vector_type(8)))  float  v8f;

__device__ __forceinline__ v8f wmma_bf16(v16bf a, v16bf b, v8f c) {
  // D = A(16x32) * B(32x16) + C, f32 accum
  return __builtin_amdgcn_wmma_f32_16x16x32_bf16(false, a, false, b, (short)0, c, false, false);
}

__device__ __forceinline__ float sigm(float x) { return 1.0f / (1.0f + __expf(-x)); }

// ---------------------------------------------------------------- converts
__global__ __launch_bounds__(256) void f32_to_bf16_k(const float* __restrict__ s,
                                                     __bf16* __restrict__ d, int n) {
  int i = blockIdx.x * 256 + threadIdx.x;
  if (i < n) d[i] = (__bf16)s[i];
}

// ------------------------------------------------- embedding gather + mask
__global__ __launch_bounds__(256) void embed_mask_k(const int* __restrict__ x,
                                                    const int* __restrict__ lengths,
                                                    const float* __restrict__ emb,
                                                    __bf16* __restrict__ wv) {
  int i = blockIdx.x * 256 + threadIdx.x;          // 0 .. B*N*D-1 (262144)
  int row = i >> 9;                                // b*N + t
  int dcol = i & 511;
  int b = row >> 5, t = row & 31;
  float v = (t < lengths[b]) ? emb[(size_t)x[row] * D_ + dcol] : 0.0f;
  wv[i] = (__bf16)v;
}

// ------------------------------------------- generic WMMA GEMM: C = A*W^T(+bias)
// A: [M][K] bf16 row-major, W: [Nc][K] bf16 row-major, C: [M][Nc] f32
// grid.x = M/16, grid.y = Nc/128; 8 waves, one 16x16 tile per wave.
// A panel (shared by all 8 waves) is staged once into LDS to kill the 8x
// redundant L2 reads; W streams from L2 with prefetch 8 k-steps (256 elements) ahead.
__global__ __launch_bounds__(256) void gemm_bf16_wmma(const __bf16* __restrict__ A,
                                                      const __bf16* __restrict__ W,
                                                      const float* __restrict__ bias,
                                                      float* __restrict__ C,
                                                      int K, int Nc) {
  __shared__ __bf16 Alds[16 * 1024];               // up to K=1024 (32 KB)

  int wave = threadIdx.x >> 5, lane = threadIdx.x & 31;
  int mt = blockIdx.x;
  int nt = blockIdx.y * 8 + wave;
  int lm = lane & 15, lk = lane >> 4;
  int bn = nt * 16 + lm;                           // W row (== B column) for this lane

  // cooperative flat copy of the 16xK A panel (rows are contiguous in global)
  const __bf16* Ablk = A + (size_t)mt * 16 * K;
  for (int i = threadIdx.x * 8; i < 16 * K; i += 256 * 8)
    *(uint4*)&Alds[i] = *(const uint4*)(Ablk + i);
  __syncthreads();

  const __bf16* wrow = W + (size_t)bn * K;
  v8f acc = {0.f,0.f,0.f,0.f,0.f,0.f,0.f,0.f};
  for (int kk = 0; kk < K; kk += 32) {
    __builtin_prefetch(wrow + kk + 256, 0, 3);     // 8 k-steps (256 elems) ahead -> global_prefetch_b8
    v16bf a = *(const v16bf*)&Alds[lm * K + kk + lk * 16];
    v16bf b = *(const v16bf*)(wrow + kk + lk * 16);
    acc = wmma_bf16(a, b, acc);
  }
  float bv = bias ? bias[bn] : 0.0f;
  int mrow = mt * 16 + (lk << 3);
#pragma unroll
  for (int r = 0; r < 8; ++r)
    C[(size_t)(mrow + r) * Nc + bn] = acc[r] + bv;
}

// ----------------------------------------------------- fused span LSTM recurrence
// grid.x enumerates (k, tile) over all span lengths: sum_k (32-k) = 496 blocks.
// grid.y = direction (0=fwd, 1=bwd). Block = 16 sequences, 8 waves.
// Wave w owns h-columns [w*64, w*64+64): computes gates i,f,g,o for them via WMMA
// (A = h in LDS, shared; B = W_hh rows), keeps c and hsum in registers.
__global__ __launch_bounds__(256) void span_lstm_k(const float* __restrict__ pre_f,
                                                   const float* __restrict__ pre_b,
                                                   const __bf16* __restrict__ Whf,
                                                   const __bf16* __restrict__ Whb,
                                                   __bf16* __restrict__ hsum) {
  // decode (k, tile) from blockIdx.x
  int bid = blockIdx.x, k = 1, tile = bid;
  for (int kk = 1; kk < 32; ++kk) {
    int cnt = 32 - kk;
    if (tile < cnt) { k = kk; break; }
    tile -= cnt;
  }
  const int Nk = 32 - k;
  const int dir = blockIdx.y;
  const float*  pre = dir ? pre_b : pre_f;
  const __bf16* Wh  = dir ? Whb  : Whf;

  int wave = threadIdx.x >> 5, lane = threadIdx.x & 31;
  int lm = lane & 15, lk = lane >> 4;
  int hbase = wave * 64;
  int mbase = lk << 3;

  __shared__ __bf16 h_lds[16][H_];           // 16 KB, shared A operand
  for (int i = threadIdx.x; i < 16 * H_; i += 256) ((__bf16*)h_lds)[i] = (__bf16)0.0f;

  // token-row base per accumulator register row
  int baserow[8];
#pragma unroll
  for (int r = 0; r < 8; ++r) {
    int m = mbase + r;
    int q = tile * 16 + m;                   // global sequence id for this k
    int b = q / Nk, s = q - b * Nk;          // batch, span start
    baserow[r] = b * N_ + s;
  }

  float creg[4][8], hsreg[4][8];
#pragma unroll
  for (int nt = 0; nt < 4; ++nt)
#pragma unroll
    for (int r = 0; r < 8; ++r) { creg[nt][r] = 0.0f; hsreg[nt][r] = 0.0f; }

  __syncthreads();

  for (int p = 0; p <= k; ++p) {
    int toff = dir ? (k - p) : p;            // bwd walks the span reversed

    // prefetch this step's gate preactivations so the L2->L0 fill overlaps
    // with the 256 WMMAs below (emits global_prefetch_b8)
#pragma unroll
    for (int r = 0; r < 8; ++r) {
      const float* prow = pre + (size_t)(baserow[r] + toff) * G4H;
#pragma unroll
      for (int g = 0; g < 4; ++g)
        __builtin_prefetch(prow + g * H_ + hbase + lm, 0, 3);
    }

    const v8f vzero = {0.f,0.f,0.f,0.f,0.f,0.f,0.f,0.f};
    v8f acc[4][4];
#pragma unroll
    for (int g = 0; g < 4; ++g)
#pragma unroll
      for (int nt = 0; nt < 4; ++nt) acc[g][nt] = vzero;

    // g[16 x 256] = h[16 x 512] @ Wh^T  (this wave's 4 gates x 4 N-tiles)
    for (int kk = 0; kk < H_; kk += 32) {
      v16bf a = *(const v16bf*)&h_lds[lm][kk + lk * 16];
#pragma unroll
      for (int g = 0; g < 4; ++g) {
#pragma unroll
        for (int nt = 0; nt < 4; ++nt) {
          int gcol = g * H_ + hbase + nt * 16 + lm;
          v16bf b = *(const v16bf*)(Wh + (size_t)gcol * H_ + kk + lk * 16);
          acc[g][nt] = wmma_bf16(a, b, acc[g][nt]);
        }
      }
    }
    __syncthreads();                         // all h reads done before update

#pragma unroll
    for (int r = 0; r < 8; ++r) {
      const float* prow = pre + (size_t)(baserow[r] + toff) * G4H;
      int m = mbase + r;
#pragma unroll
      for (int nt = 0; nt < 4; ++nt) {
        int col = hbase + nt * 16 + lm;
        float iv = sigm(acc[0][nt][r] + prow[col]);
        float fv = sigm(acc[1][nt][r] + prow[H_ + col]);
        float gv = tanhf(acc[2][nt][r] + prow[2 * H_ + col]);
        float ov = sigm(acc[3][nt][r] + prow[3 * H_ + col]);
        float c = fv * creg[nt][r] + iv * gv;
        creg[nt][r] = c;
        float h = ov * tanhf(c);
        hsreg[nt][r] += h;
        h_lds[m][col] = (__bf16)h;
      }
    }
    __syncthreads();
  }

  // hsum rows follow output ordering: row = b*496 + spanoff(k) + s
  int spanoff = ((k - 1) * (2 * N_ - k)) >> 1;
#pragma unroll
  for (int r = 0; r < 8; ++r) {
    int m = mbase + r;
    int q = tile * 16 + m;
    int b = q / Nk, s = q - b * Nk;
    int grow = b * SPANS + spanoff + s;
#pragma unroll
    for (int nt = 0; nt < 4; ++nt) {
      int col = hbase + nt * 16 + lm;
      hsum[(size_t)grow * (2 * H_) + dir * H_ + col] = (__bf16)hsreg[nt][r];
    }
  }
}

// ---------------------------------------- (k+1)*b_out bias + l2-norm over SEM
__global__ __launch_bounds__(256) void bias_l2norm_k(float* __restrict__ out,
                                                     const float* __restrict__ b_out) {
  int wave = threadIdx.x >> 5, lane = threadIdx.x & 31;
  int row = blockIdx.x * 8 + wave;            // of B*SPANS*NT rows of SEM
  int rem = row % (SPANS * NT_);
  int off = rem >> 4, ntc = rem & 15;
  int cum = 0, k = 1;
  for (int kk = 1; kk < 32; ++kk) {
    int cnt = 32 - kk;
    if (off < cum + cnt) { k = kk; break; }
    cum += cnt;
  }
  float kb = (float)(k + 1);
  float* rp = out + (size_t)row * SEM_;
  const float* bp = b_out + ntc * SEM_;
  float v[8], ss = 0.0f;
#pragma unroll
  for (int i = 0; i < 8; ++i) {
    int c = lane + i * 32;
    v[i] = rp[c] + kb * bp[c];
    ss += v[i] * v[i];
  }
#pragma unroll
  for (int m = 16; m >= 1; m >>= 1) ss += __shfl_xor(ss, m, 32);
  float inv = rsqrtf(ss);
#pragma unroll
  for (int i = 0; i < 8; ++i) rp[lane + i * 32] = v[i] * inv;
}

// ------------------------------------------------------------------- launch
extern "C" void kernel_launch(void* const* d_in, const int* in_sizes, int n_in,
                              void* d_out, int out_size, void* d_ws, size_t ws_size,
                              hipStream_t stream) {
  (void)in_sizes; (void)n_in; (void)out_size; (void)ws_size;
  const int*   x       = (const int*)d_in[0];
  const int*   lengths = (const int*)d_in[1];
  const float* emb     = (const float*)d_in[2];
  const float* Wih_f   = (const float*)d_in[3];
  const float* Whh_f   = (const float*)d_in[4];
  const float* b_f     = (const float*)d_in[5];
  const float* Wih_b   = (const float*)d_in[6];
  const float* Whh_b   = (const float*)d_in[7];
  const float* b_b     = (const float*)d_in[8];
  const float* W_out   = (const float*)d_in[9];
  const float* b_out   = (const float*)d_in[10];
  float* out = (float*)d_out;

  char* ws = (char*)d_ws;
  size_t off = 0;
  auto alloc = [&](size_t bytes) -> void* {
    void* p = ws + off;
    off = (off + bytes + 255) & ~(size_t)255;
    return p;
  };
  __bf16* wv16    = (__bf16*)alloc((size_t)B_ * N_ * D_ * 2);       // 512 KB
  __bf16* Wihf16  = (__bf16*)alloc((size_t)G4H * D_ * 2);           // 2 MB
  __bf16* Wihb16  = (__bf16*)alloc((size_t)G4H * D_ * 2);
  __bf16* Whhf16  = (__bf16*)alloc((size_t)G4H * H_ * 2);
  __bf16* Whhb16  = (__bf16*)alloc((size_t)G4H * H_ * 2);
  __bf16* Wout16  = (__bf16*)alloc((size_t)OUTD * 2 * H_ * 2);      // 8 MB
  float*  pre_f   = (float*) alloc((size_t)B_ * N_ * G4H * 4);      // 4 MB
  float*  pre_b   = (float*) alloc((size_t)B_ * N_ * G4H * 4);
  __bf16* hsum16  = (__bf16*)alloc((size_t)ROWS * 2 * H_ * 2);      // 16 MB

  // 1) weights -> bf16
  f32_to_bf16_k<<<(G4H * D_) / 256, 256, 0, stream>>>(Wih_f, Wihf16, G4H * D_);
  f32_to_bf16_k<<<(G4H * D_) / 256, 256, 0, stream>>>(Wih_b, Wihb16, G4H * D_);
  f32_to_bf16_k<<<(G4H * H_) / 256, 256, 0, stream>>>(Whh_f, Whhf16, G4H * H_);
  f32_to_bf16_k<<<(G4H * H_) / 256, 256, 0, stream>>>(Whh_b, Whhb16, G4H * H_);
  f32_to_bf16_k<<<(OUTD * 2 * H_) / 256, 256, 0, stream>>>(W_out, Wout16, OUTD * 2 * H_);

  // 2) masked embeddings (bf16)
  embed_mask_k<<<(B_ * N_ * D_) / 256, 256, 0, stream>>>(x, lengths, emb, wv16);

  // 3) per-token gate preactivations: pre = wv @ W_ih^T + b   [512 x 2048]
  gemm_bf16_wmma<<<dim3((B_ * N_) / 16, G4H / 128), 256, 0, stream>>>(
      wv16, Wihf16, b_f, pre_f, D_, G4H);
  gemm_bf16_wmma<<<dim3((B_ * N_) / 16, G4H / 128), 256, 0, stream>>>(
      wv16, Wihb16, b_b, pre_b, D_, G4H);

  // 4) all span lengths + both directions in one launch (496 x 2 blocks)
  span_lstm_k<<<dim3(SPANS, 2), 256, 0, stream>>>(pre_f, pre_b, Whhf16, Whhb16, hsum16);

  // 5) projection: out_raw = hsum @ W_out^T   [7936 x 4096]
  gemm_bf16_wmma<<<dim3(ROWS / 16, OUTD / 128), 256, 0, stream>>>(
      hsum16, Wout16, nullptr, out, 2 * H_, OUTD);

  // 6) (k+1)*b_out bias + l2norm over SEM, in place
  bias_l2norm_k<<<(B_ * SPANS * NT_) / 8, 256, 0, stream>>>(out, b_out);
}
